// HGTLayer_39367670235358
// MI455X (gfx1250) — matrix operand order repaired and checked by reading, not compiled, not run on hardware
//
#include <hip/hip_runtime.h>
#include <hip/hip_bf16.h>
#include <math.h>

typedef __attribute__((ext_vector_type(16))) _Float16 v16h;
typedef __attribute__((ext_vector_type(8)))  _Float16 v8h;
typedef __attribute__((ext_vector_type(4)))  _Float16 v4h;
typedef __attribute__((ext_vector_type(8)))  float    v8f;

// Problem constants (match reference)
constexpr int NN   = 32768;
constexpr int EE   = 120000;
constexpr int INF  = 512;
constexpr int OUTF = 512;
constexpr int HH   = 8;
constexpr int DKK  = 64;

// ---------------------------------------------------------------------------
// small helpers
// ---------------------------------------------------------------------------
__device__ inline float gelu_exact(float x) {
    return 0.5f * x * (1.0f + erff(x * 0.70710678118654752f));
}

__device__ inline void atomicMaxF(float* addr, float val) {
    int* ia = (int*)addr;
    int old = __float_as_int(*addr);
    while (__int_as_float(old) < val) {
        int assumed = old;
        old = atomicCAS(ia, assumed, __float_as_int(val));
        if (old == assumed) break;
    }
}

__global__ void fillf_kernel(float* __restrict__ p, float v, size_t n) {
    size_t i = (size_t)blockIdx.x * blockDim.x + threadIdx.x;
    if (i < n) p[i] = v;
}

// ---------------------------------------------------------------------------
// Weight folding: Weff[z] = W[st] @ BlockDiag(R[r]);  z = r + 3*isV
// ---------------------------------------------------------------------------
__global__ void fold_w_kernel(const float* __restrict__ Wk, const float* __restrict__ Wv,
                              const float* __restrict__ Ratt, const float* __restrict__ Rmsg,
                              float* __restrict__ Weff) {
    const int z   = blockIdx.z;      // 0..5
    const int r   = z % 3;
    const int isV = z / 3;
    const int stArr[3] = {0, 1, 0};
    const int st = stArr[r];
    const float* W = (isV ? Wv : Wk) + (size_t)st * INF * OUTF;
    const float* R = (isV ? Rmsg : Ratt) + (size_t)r * HH * DKK * DKK;

    int idx = blockIdx.x * blockDim.x + threadIdx.x;     // 0 .. 512*512-1
    int row = idx >> 9;
    int col = idx & 511;
    int h   = col >> 6;
    int cj  = col & 63;
    const float* Wr = W + (size_t)row * OUTF + h * DKK;
    const float* Rh = R + (size_t)h * DKK * DKK + cj;
    float acc = 0.f;
#pragma unroll 8
    for (int d = 0; d < DKK; ++d) acc += Wr[d] * Rh[(size_t)d * DKK];
    Weff[(size_t)z * INF * OUTF + idx] = acc;
}

__global__ void fold_b_kernel(const float* __restrict__ bk, const float* __restrict__ bv,
                              const float* __restrict__ Ratt, const float* __restrict__ Rmsg,
                              float* __restrict__ beff) {
    int idx = blockIdx.x * blockDim.x + threadIdx.x;     // 0 .. 6*512-1
    if (idx >= 6 * OUTF) return;
    const int z   = idx >> 9;
    const int col = idx & 511;
    const int r   = z % 3;
    const int isV = z / 3;
    const int stArr[3] = {0, 1, 0};
    const float* b = (isV ? bv : bk) + stArr[r] * OUTF;
    const float* R = (isV ? Rmsg : Ratt) + (size_t)r * HH * DKK * DKK
                     + (size_t)(col >> 6) * DKK * DKK + (col & 63);
    float acc = 0.f;
#pragma unroll 8
    for (int d = 0; d < DKK; ++d) acc += b[(col >> 6) * DKK + d] * R[(size_t)d * DKK];
    beff[idx] = acc;
}

// ---------------------------------------------------------------------------
// Tiled WMMA GEMM:  C[M x Nc] = op(A[M x K]) * B[K x Nc] + bias
//   Block = 256 threads = 8 waves -> 128x64 C tile (waves 4x2, each wave a
//   2x2 grid of 16x16 accumulators).  A staged row-major (K-contiguous), B
//   staged TRANSPOSED (K-contiguous) so every WMMA fragment is two aligned
//   16B LDS loads (ds_load_b128), no cross-lane packing.
//   GELU:   apply exact GELU to A elements while staging.
//   OMODE 0: out is _Float16*, write acc+bias.
//   OMODE 1: out is float*, write (acc+bias)*alpha + resid*(1-alpha).
// ---------------------------------------------------------------------------
template <bool GELU, int OMODE>
__global__ __launch_bounds__(256)
void gemm_wmma_kernel(const float* __restrict__ A, const float* __restrict__ B,
                      const float* __restrict__ bias, void* __restrict__ out,
                      const float* __restrict__ resid, const float* __restrict__ skipPtr,
                      int M, int Nc, int K) {
    // rows padded to 40 halves (80B = 20 banks): 16B aligned, conflict-free
    __shared__ _Float16 As[128][40];   // A tile: 128 rows x 32 K
    __shared__ _Float16 Bt[64][40];    // B tile transposed: 64 cols x 32 K

    const int tid  = threadIdx.x;
    const int lane = tid & 31;
    const int w    = tid >> 5;        // wave id 0..7
    const int wrow = w >> 1;          // 0..3  -> M offset 32*wrow
    const int wcol = w & 1;           // 0..1  -> N offset 32*wcol

    const int m0 = blockIdx.y * 128;
    const int n0 = blockIdx.x * 64;

    v8f c00 = {}, c01 = {}, c10 = {}, c11 = {};

    for (int k0 = 0; k0 < K; k0 += 32) {
        __syncthreads();
        // ---- stage A tile (128 rows x 32 K) as fp16, K-contiguous ----
#pragma unroll
        for (int it = 0; it < 4; ++it) {
            int f   = tid + it * 256;           // float4 index, 0..1023
            int row = f >> 3;
            int col = (f & 7) * 4;
            float4 x = *(const float4*)(A + (size_t)(m0 + row) * K + k0 + col);
            if (GELU) {
                x.x = gelu_exact(x.x); x.y = gelu_exact(x.y);
                x.z = gelu_exact(x.z); x.w = gelu_exact(x.w);
            }
            v4h hx = { (_Float16)x.x, (_Float16)x.y, (_Float16)x.z, (_Float16)x.w };
            *(v4h*)&As[row][col] = hx;
        }
        // ---- stage B tile (32 K x 64 N) transposed into Bt[n][k] ----
        // global read: col = tid&63 (coalesced), 4 consecutive K rows
#pragma unroll
        for (int it = 0; it < 2; ++it) {
            int f    = tid + it * 256;          // 0..511
            int col  = f & 63;
            int row4 = (f >> 6) * 4;            // 0,4,...,28
            const float* bp = B + (size_t)(k0 + row4) * Nc + n0 + col;
            v4h hx;
            hx[0] = (_Float16)bp[0];
            hx[1] = (_Float16)bp[(size_t)Nc];
            hx[2] = (_Float16)bp[(size_t)2 * Nc];
            hx[3] = (_Float16)bp[(size_t)3 * Nc];
            *(v4h*)&Bt[col][row4] = hx;
        }
        __syncthreads();

        // ---- fragments: all K-contiguous -> aligned 16B LDS loads ----
        // A (16-bit 16x32): lanes 0-15 K=0..7 & 16..23; lanes 16-31 K=8..15 & 24..31
        const int am = lane & 15;
        const int kb = (lane >> 4) * 8;
        // B (32x16): lane&15 = column; lanes 0-15 K=0..7 & 16..23; lanes 16-31 K=8..15 & 24..31
        const int bn = lane & 15;
        const int kOff = (lane >> 4) * 8;

        v8h a0lo = *(const v8h*)&As[wrow * 32 + am][kb];
        v8h a0hi = *(const v8h*)&As[wrow * 32 + am][16 + kb];
        v8h a1lo = *(const v8h*)&As[wrow * 32 + 16 + am][kb];
        v8h a1hi = *(const v8h*)&As[wrow * 32 + 16 + am][16 + kb];
        v8h b0lo = *(const v8h*)&Bt[wcol * 32 + bn][kOff];
        v8h b0hi = *(const v8h*)&Bt[wcol * 32 + bn][kOff + 16];
        v8h b1lo = *(const v8h*)&Bt[wcol * 32 + 16 + bn][kOff];
        v8h b1hi = *(const v8h*)&Bt[wcol * 32 + 16 + bn][kOff + 16];

        v16h a0, a1, b0, b1;
#pragma unroll
        for (int j = 0; j < 8; ++j) {
            a0[j] = a0lo[j]; a0[j + 8] = a0hi[j];
            a1[j] = a1lo[j]; a1[j + 8] = a1hi[j];
            b0[j] = b0lo[j]; b0[j + 8] = b0hi[j];
            b1[j] = b1lo[j]; b1[j + 8] = b1hi[j];
        }
        c00 = __builtin_amdgcn_wmma_f32_16x16x32_f16(false, a0, false, b0, (short)0, c00, false, false);
        c01 = __builtin_amdgcn_wmma_f32_16x16x32_f16(false, a0, false, b1, (short)0, c01, false, false);
        c10 = __builtin_amdgcn_wmma_f32_16x16x32_f16(false, a1, false, b0, (short)0, c10, false, false);
        c11 = __builtin_amdgcn_wmma_f32_16x16x32_f16(false, a1, false, b1, (short)0, c11, false, false);
    }

    float alpha = 0.f, oneMinus = 0.f;
    if (OMODE == 1) {
        float sv = skipPtr[0];
        alpha    = 1.f / (1.f + __expf(-sv));
        oneMinus = 1.f - alpha;
    }

    // C/D layout: VGPR i -> row i (lanes 0-15) / row 8+i (lanes 16-31), col = lane&15
    const int rb0 = m0 + wrow * 32 + ((lane >> 4) << 3);
    const int c0  = n0 + wcol * 32 + (lane & 15);
#pragma unroll
    for (int i = 0; i < 8; ++i) {
        const int gm0 = rb0 + i;        // tiles c00 / c01
        const int gm1 = gm0 + 16;       // tiles c10 / c11
        float v00 = c00[i] + bias[c0];
        float v01 = c01[i] + bias[c0 + 16];
        float v10 = c10[i] + bias[c0];
        float v11 = c11[i] + bias[c0 + 16];
        if (OMODE == 0) {
            _Float16* o = (_Float16*)out;
            o[(size_t)gm0 * Nc + c0]      = (_Float16)v00;
            o[(size_t)gm0 * Nc + c0 + 16] = (_Float16)v01;
            o[(size_t)gm1 * Nc + c0]      = (_Float16)v10;
            o[(size_t)gm1 * Nc + c0 + 16] = (_Float16)v11;
        } else {
            float* o  = (float*)out;
            size_t i0 = (size_t)gm0 * Nc + c0;
            size_t i1 = (size_t)gm1 * Nc + c0;
            o[i0]      = v00 * alpha + resid[i0] * oneMinus;
            o[i0 + 16] = v01 * alpha + resid[i0 + 16] * oneMinus;
            o[i1]      = v10 * alpha + resid[i1] * oneMinus;
            o[i1 + 16] = v11 * alpha + resid[i1 + 16] * oneMinus;
        }
    }
}

// ---------------------------------------------------------------------------
// Edge-phase kernels (one thread per (edge, head))
// ---------------------------------------------------------------------------
__global__ void edge_score_kernel(const int* __restrict__ src, const int* __restrict__ dst,
                                  const _Float16* __restrict__ q, const _Float16* __restrict__ ke,
                                  const float* __restrict__ pri,
                                  float* __restrict__ s, float* __restrict__ m, int E) {
    int idx = blockIdx.x * blockDim.x + threadIdx.x;
    if (idx >= E * HH) return;
    const int h  = idx & 7;
    const int e  = idx >> 3;
    const int sI = src[e];
    const int dI = dst[e];
    const v8h* qp = (const v8h*)(q  + (size_t)dI * OUTF + h * DKK);
    const v8h* kp = (const v8h*)(ke + (size_t)sI * OUTF + h * DKK);
    float acc = 0.f;
#pragma unroll
    for (int i = 0; i < 8; ++i) {
        v8h qa = qp[i], ka = kp[i];
#pragma unroll
        for (int j = 0; j < 8; ++j) acc += (float)qa[j] * (float)ka[j];
    }
    float sv = acc * pri[h] * 0.125f;    // 1/sqrt(64)
    s[idx] = sv;
    atomicMaxF(m + (size_t)dI * HH + h, sv);
}

__global__ void edge_exp_kernel(const int* __restrict__ dst,
                                const float* __restrict__ s, const float* __restrict__ m,
                                float* __restrict__ exb, float* __restrict__ den, int E) {
    int idx = blockIdx.x * blockDim.x + threadIdx.x;
    if (idx >= E * HH) return;
    const int h  = idx & 7;
    const int e  = idx >> 3;
    const int dI = dst[e];
    float ex = __expf(s[idx] - m[(size_t)dI * HH + h]);
    exb[idx] = ex;
    atomicAdd(den + (size_t)dI * HH + h, ex);
}

__global__ void edge_scatter_kernel(const int* __restrict__ src, const int* __restrict__ dst,
                                    const _Float16* __restrict__ ve,
                                    const float* __restrict__ exb, const float* __restrict__ den,
                                    float* __restrict__ agg, int E) {
    int idx = blockIdx.x * blockDim.x + threadIdx.x;
    if (idx >= E * HH) return;
    const int h  = idx & 7;
    const int e  = idx >> 3;
    const int sI = src[e];
    const int dI = dst[e];
    float attn = exb[idx] / den[(size_t)dI * HH + h];
    const v8h* vp = (const v8h*)(ve + (size_t)sI * OUTF + h * DKK);
    float* ap = agg + (size_t)dI * OUTF + h * DKK;
#pragma unroll
    for (int i = 0; i < 8; ++i) {
        v8h va = vp[i];
#pragma unroll
        for (int j = 0; j < 8; ++j) atomicAdd(ap + i * 8 + j, (float)va[j] * attn);
    }
}

// ---------------------------------------------------------------------------
// launch
// ---------------------------------------------------------------------------
extern "C" void kernel_launch(void* const* d_in, const int* in_sizes, int n_in,
                              void* d_out, int out_size, void* d_ws, size_t ws_size,
                              hipStream_t stream) {
    const float* h0 = (const float*)d_in[0];
    const float* h1 = (const float*)d_in[1];
    const float* hs[2] = {h0, h1};
    const int* srcs[3] = {(const int*)d_in[2], (const int*)d_in[4], (const int*)d_in[6]};
    const int* dsts[3] = {(const int*)d_in[3], (const int*)d_in[5], (const int*)d_in[7]};
    const float* Wk = (const float*)d_in[8];
    const float* bk = (const float*)d_in[9];
    const float* Wq = (const float*)d_in[10];
    const float* bq = (const float*)d_in[11];
    const float* Wv = (const float*)d_in[12];
    const float* bv = (const float*)d_in[13];
    const float* Wa = (const float*)d_in[14];
    const float* ba = (const float*)d_in[15];
    const float* rel_pri = (const float*)d_in[16];
    const float* rel_att = (const float*)d_in[17];
    const float* rel_msg = (const float*)d_in[18];
    const float* skip    = (const float*)d_in[19];
    float* out = (float*)d_out;

    const int relSrcT[3] = {0, 1, 0};
    const int relDstT[3] = {0, 0, 1};

    // workspace carve-up (256B aligned)
    char* wsp = (char*)d_ws;
    auto carve = [&](size_t bytes) -> char* {
        char* p = wsp;
        wsp += (bytes + 255) & ~(size_t)255;
        return p;
    };
    const size_t nodeMatH = (size_t)NN * OUTF * sizeof(_Float16);
    const size_t nodeMatF = (size_t)NN * OUTF * sizeof(float);
    _Float16* qbuf  = (_Float16*)carve(2 * nodeMatH);
    _Float16* kebuf = (_Float16*)carve(3 * nodeMatH);
    _Float16* vebuf = (_Float16*)carve(3 * nodeMatH);
    float*    Weff  = (float*)carve(6 * (size_t)INF * OUTF * sizeof(float));
    float*    beff  = (float*)carve(6 * (size_t)OUTF * sizeof(float));
    float*    aggb  = (float*)carve(2 * nodeMatF);
    float*    sbuf  = (float*)carve((size_t)EE * HH * sizeof(float));
    float*    exbuf = (float*)carve((size_t)EE * HH * sizeof(float));
    float*    mbuf  = (float*)carve((size_t)NN * HH * sizeof(float));
    float*    denb  = (float*)carve((size_t)NN * HH * sizeof(float));

    // 1) fold per-relation head transforms into projection weights
    fold_w_kernel<<<dim3((INF * OUTF) / 256, 1, 6), 256, 0, stream>>>(Wk, Wv, rel_att, rel_msg, Weff);
    fold_b_kernel<<<dim3((6 * OUTF + 255) / 256), 256, 0, stream>>>(bk, bv, rel_att, rel_msg, beff);

    const dim3 gemmGrid(OUTF / 64, NN / 128);
    // 2) q projections (fp16 out)
    for (int t = 0; t < 2; ++t) {
        gemm_wmma_kernel<false, 0><<<gemmGrid, 256, 0, stream>>>(
            hs[t], Wq + (size_t)t * INF * OUTF, bq + t * OUTF,
            (void*)(qbuf + (size_t)t * NN * OUTF), nullptr, nullptr, NN, OUTF, INF);
    }
    // 3) ke / ve via folded weights (fp16 out)
    for (int r = 0; r < 3; ++r) {
        gemm_wmma_kernel<false, 0><<<gemmGrid, 256, 0, stream>>>(
            hs[relSrcT[r]], Weff + (size_t)r * INF * OUTF, beff + r * OUTF,
            (void*)(kebuf + (size_t)r * NN * OUTF), nullptr, nullptr, NN, OUTF, INF);
        gemm_wmma_kernel<false, 0><<<gemmGrid, 256, 0, stream>>>(
            hs[relSrcT[r]], Weff + (size_t)(3 + r) * INF * OUTF, beff + (3 + r) * OUTF,
            (void*)(vebuf + (size_t)r * NN * OUTF), nullptr, nullptr, NN, OUTF, INF);
    }

    // 4) zero aggregation buffers
    {
        size_t n = 2 * (size_t)NN * OUTF;
        fillf_kernel<<<(unsigned)((n + 255) / 256), 256, 0, stream>>>(aggb, 0.f, n);
    }

    // 5) per-relation edge softmax + weighted scatter
    const int ehThreads = EE * HH;
    const dim3 eGrid((ehThreads + 255) / 256);
    for (int r = 0; r < 3; ++r) {
        const int dt = relDstT[r];
        size_t nm = (size_t)NN * HH;
        fillf_kernel<<<(unsigned)((nm + 255) / 256), 256, 0, stream>>>(mbuf, -3.0e38f, nm);
        fillf_kernel<<<(unsigned)((nm + 255) / 256), 256, 0, stream>>>(denb, 0.f, nm);
        edge_score_kernel<<<eGrid, 256, 0, stream>>>(
            srcs[r], dsts[r],
            qbuf + (size_t)dt * NN * OUTF, kebuf + (size_t)r * NN * OUTF,
            rel_pri + r * HH, sbuf, mbuf, EE);
        edge_exp_kernel<<<eGrid, 256, 0, stream>>>(dsts[r], sbuf, mbuf, exbuf, denb, EE);
        edge_scatter_kernel<<<eGrid, 256, 0, stream>>>(
            srcs[r], dsts[r], vebuf + (size_t)r * NN * OUTF,
            exbuf, denb, aggb + (size_t)dt * NN * OUTF, EE);
    }

    // 6) output: gelu(agg) @ Wa + ba, skip-blend with h  (fused epilogue)
    for (int t = 0; t < 2; ++t) {
        gemm_wmma_kernel<true, 1><<<gemmGrid, 256, 0, stream>>>(
            aggb + (size_t)t * NN * OUTF, Wa + (size_t)t * OUTF * OUTF, ba + t * OUTF,
            (void*)(out + (size_t)t * NN * OUTF), hs[t], skip + t, NN, OUTF, OUTF);
    }
}
